// ClusterModelGGMA_1245540516364
// MI455X (gfx1250) — matrix operand order repaired and checked by reading, not compiled, not run on hardware
//
#include <hip/hip_runtime.h>
#include <hip/hip_bf16.h>
#include <math.h>

// ---------------------------------------------------------------------------
// Types for CDNA5 WMMA (gfx1250, wave32)
// ---------------------------------------------------------------------------
typedef __bf16 bf16_t;
typedef __attribute__((ext_vector_type(16))) __bf16 v16bf;
typedef __attribute__((ext_vector_type(8)))  float  v8f;
typedef unsigned int v4ui __attribute__((ext_vector_type(4)));
typedef int          v8si __attribute__((ext_vector_type(8)));
typedef int          v4si __attribute__((ext_vector_type(4)));

#define DEV __device__ __forceinline__

DEV v8f wmma_bf16(v16bf a, v16bf b, v8f c) {
  return __builtin_amdgcn_wmma_f32_16x16x32_bf16(
      false, a, false, b, (short)0, c, false, false);
}

// A-matrix 16x32 bf16 fragment, row-major source, row stride ld (elements).
// Per-lane: one contiguous 32B run (k = 16*half + s).
DEV v16bf load_a_frag(const bf16_t* p, int ld) {
  int lane = threadIdx.x & 31;
  int m = lane & 15, half = lane >> 4;
  v16bf a;
#pragma unroll
  for (int v = 0; v < 8; ++v) {
    int k0 = (half << 4) + (v << 1);
    const bf16_t* q = p + (size_t)m * ld + k0;
    a[2 * v]     = q[0];
    a[2 * v + 1] = q[1];
  }
  return a;
}

// B-matrix 32x16 fragment where memory holds B^T row-major:
//   B[k][n] = src[n*ld + k].  Per-lane: one contiguous 32B run.
DEV v16bf load_bT_frag(const bf16_t* p, int ld) {
  int lane = threadIdx.x & 31;
  int n = lane & 15, half = lane >> 4;
  const bf16_t* q = p + (size_t)n * ld + (half << 4);
  v16bf b;
#pragma unroll
  for (int v = 0; v < 16; ++v) b[v] = q[v];
  return b;
}

// C/D 16x16 f32: N=lane&15, M=(lane>>4)*8 + r
DEV void store_c_f32(float* p, int ld, v8f c) {
  int lane = threadIdx.x & 31;
  int n = lane & 15, mb = (lane >> 4) << 3;
#pragma unroll
  for (int r = 0; r < 8; ++r) p[(size_t)(mb + r) * ld + n] = c[r];
}

DEV float block_reduce_sum256(float v, float* red) {
  int tid = threadIdx.x;
  red[tid] = v;
  __syncthreads();
  for (int off = 128; off > 0; off >>= 1) {
    if (tid < off) red[tid] += red[tid + off];
    __syncthreads();
  }
  float r = red[0];
  __syncthreads();
  return r;
}

// ---------------------------------------------------------------------------
// Model constants
// ---------------------------------------------------------------------------
static constexpr int SEQS  = 16;
static constexpr int SLEN  = 640;
static constexpr int DM    = 768;
static constexpr int DF    = 3072;
static constexpr int NH    = 12;
static constexpr int HD    = 64;
static constexpr int MROWS = SEQS * SLEN;  // 10240

// ---------------------------------------------------------------------------
// Tensor Data Mover: 2D bf16 tile (tile_d0 x tile_d1) global -> LDS with
// 8-element row padding (pad_interval=16 DW, pad_amount=4 DW).
// ---------------------------------------------------------------------------
DEV void tdm_load_tile_bf16(unsigned lds_byte_addr, const bf16_t* gsrc,
                            unsigned tensor_d0, unsigned tensor_d1,
                            unsigned long long d0_stride_elems,
                            unsigned tile_d0, unsigned tile_d1) {
  unsigned long long ga = (unsigned long long)(size_t)gsrc;
  v4ui g0;
  g0[0] = 1u;                                   // count=1, user mode
  g0[1] = lds_byte_addr;                        // lds_addr
  g0[2] = (unsigned)(ga & 0xFFFFFFFFu);         // global_addr[31:0]
  g0[3] = (unsigned)((ga >> 32) & 0x1FFFFFFu) | (2u << 30);  // addr[56:32]|type=2
  v8si g1;
  g1[0] = (int)((1u << 16) | (1u << 20) | (3u << 22) | (3u << 25));
  //            data_size=2B  pad_en      ivl=16DW      amt=4DW
  g1[1] = (int)((tensor_d0 & 0xFFFFu) << 16);
  g1[2] = (int)((tensor_d0 >> 16) | ((tensor_d1 & 0xFFFFu) << 16));
  g1[3] = (int)((tensor_d1 >> 16) | (tile_d0 << 16));
  g1[4] = (int)(tile_d1 & 0xFFFFu);
  g1[5] = (int)(unsigned)(d0_stride_elems & 0xFFFFFFFFull);
  g1[6] = (int)(unsigned)((d0_stride_elems >> 32) & 0xFFFFu);
  g1[7] = 0;
  v4si g2 = {};
  v4si g3 = {};
#if defined(__clang_major__) && (__clang_major__ >= 23)
  v8si g4 = {};
  __builtin_amdgcn_tensor_load_to_lds(g0, g1, g2, g3, g4, 0);
#else
  __builtin_amdgcn_tensor_load_to_lds(g0, g1, g2, g3, 0);
#endif
}

// ---------------------------------------------------------------------------
// f32 -> bf16 convert
// ---------------------------------------------------------------------------
__global__ __launch_bounds__(256) void convert_bf16_kernel(
    const float* __restrict__ src, bf16_t* __restrict__ dst, long long n) {
  long long i = (long long)blockIdx.x * 256 + threadIdx.x;
  if (i < n) dst[i] = (bf16_t)src[i];
}

// ---------------------------------------------------------------------------
// Embedding: node-average + word/pos/type emb + LayerNorm -> x_f32, x_bf16
// ---------------------------------------------------------------------------
__global__ __launch_bounds__(256) void embed_kernel(
    const int* __restrict__ ids1, const int* __restrict__ ids2,
    const int* __restrict__ pos1, const int* __restrict__ pos2,
    const unsigned char* __restrict__ am1, const unsigned char* __restrict__ am2,
    const float* __restrict__ word_emb, const float* __restrict__ pos_emb,
    const float* __restrict__ type_emb,
    const float* __restrict__ g, const float* __restrict__ bta,
    float* __restrict__ xf, bf16_t* __restrict__ xb) {
  int row = blockIdx.x;
  int s = row / SLEN, t = row % SLEN;
  int pair = s >> 1;
  const int* ids = ((s & 1) ? ids2 : ids1) + (size_t)pair * SLEN;
  const int* ps  = ((s & 1) ? pos2 : pos1) + (size_t)pair * SLEN;
  const unsigned char* am =
      ((s & 1) ? am2 : am1) + ((size_t)pair * SLEN + t) * SLEN;

  __shared__ unsigned char cond[SLEN];
  __shared__ float red[256];
  __shared__ float cnt_sh, mean_sh, rstd_sh;

  int tid = threadIdx.x;
  int p_t = ps[t];
  bool node = (p_t == 0);

  float cnt = 0.f;
  for (int c = tid; c < SLEN; c += 256) {
    unsigned char ok = (ps[c] >= 2 && am[c]) ? 1 : 0;
    cond[c] = ok;
    cnt += (float)ok;
  }
  float csum = block_reduce_sum256(cnt, red);
  if (tid == 0) cnt_sh = csum;
  __syncthreads();

  float e[3];
  int d0 = tid;
  if (node) {
    float a0 = 0.f, a1 = 0.f, a2 = 0.f;
    for (int c = 0; c < SLEN; ++c) {
      if (cond[c]) {
        const float* w = word_emb + (size_t)ids[c] * DM;
        a0 += w[d0]; a1 += w[d0 + 256]; a2 += w[d0 + 512];
      }
    }
    float inv = 1.0f / (cnt_sh + 1e-10f);
    e[0] = a0 * inv; e[1] = a1 * inv; e[2] = a2 * inv;
  } else {
    const float* w = word_emb + (size_t)ids[t] * DM;
    e[0] = w[d0]; e[1] = w[d0 + 256]; e[2] = w[d0 + 512];
  }
  const float* pe = pos_emb + (size_t)p_t * DM;
  e[0] += pe[d0]       + type_emb[d0];
  e[1] += pe[d0 + 256] + type_emb[d0 + 256];
  e[2] += pe[d0 + 512] + type_emb[d0 + 512];

  float sum = block_reduce_sum256(e[0] + e[1] + e[2], red);
  if (tid == 0) mean_sh = sum * (1.0f / (float)DM);
  __syncthreads();
  float m = mean_sh;
  float v0 = e[0] - m, v1 = e[1] - m, v2 = e[2] - m;
  float var = block_reduce_sum256(v0 * v0 + v1 * v1 + v2 * v2, red);
  if (tid == 0) rstd_sh = rsqrtf(var * (1.0f / (float)DM) + 1e-5f);
  __syncthreads();
  float rs = rstd_sh;

  float* xo = xf + (size_t)row * DM;
  bf16_t* xbo = xb + (size_t)row * DM;
#pragma unroll
  for (int j = 0; j < 3; ++j) {
    int d = d0 + j * 256;
    float val = (e[j] - m) * rs * g[d] + bta[d];
    xo[d] = val;
    xbo[d] = (bf16_t)val;
  }
}

// ---------------------------------------------------------------------------
// bf16 WMMA GEMM, double-buffered LDS.
// Block tile 128x128, 8 waves (2x4), wave 64x32 (4x2 tiles).
// A tile (128x32) via TDM DMA into buffer^1 while computing buffer;
// B tile transposed into LDS [n][k] via register 4x4 transpose.
// One barrier per K-step; DMA fully overlapped with WMMA.
// Epilogue modes:
//   0: f32 row-major (+bias)
//   1: exact-GELU -> bf16 row-major (+bias)
//   2: bf16 head-pack       [seq][h][t][d] (+bias)   (Q, K)
//   3: bf16 head-pack-trans [seq][h][d][t] (+bias)   (V)
// ---------------------------------------------------------------------------
#define AS_LD 40   // 32 + 8 pad
#define BS_LD 40   // 32 + 8 pad (transposed: [n][k])

DEV void gemm_stage_b(const bf16_t* __restrict__ W, bf16_t* __restrict__ Bs,
                      int kb, int N, int n0, int kq, int nq) {
  union Pk { uint2 v; unsigned short u[4]; };
  Pk r0, r1, r2, r3;
  const bf16_t* wp = &W[(size_t)(kb + kq) * N + n0 + nq];
  r0.v = *(const uint2*)(wp);
  r1.v = *(const uint2*)(wp + N);
  r2.v = *(const uint2*)(wp + 2 * N);
  r3.v = *(const uint2*)(wp + 3 * N);
#pragma unroll
  for (int j = 0; j < 4; ++j) {
    Pk o;
    o.u[0] = r0.u[j]; o.u[1] = r1.u[j];
    o.u[2] = r2.u[j]; o.u[3] = r3.u[j];
    *(uint2*)&Bs[(size_t)(nq + j) * BS_LD + kq] = o.v;
  }
}

__global__ __launch_bounds__(256) void gemm_bf16_kernel(
    const bf16_t* __restrict__ A, const bf16_t* __restrict__ W,
    const float* __restrict__ bias,
    float* __restrict__ outF, bf16_t* __restrict__ outB,
    int M, int N, int K, int mode) {
  __shared__ bf16_t As[2][128 * AS_LD];
  __shared__ bf16_t Bs[2][128 * BS_LD];

  int tid = threadIdx.x;
  int n0 = blockIdx.x * 128;
  int m0 = blockIdx.y * 128;
  int wid = tid >> 5;
  int m_wave = (wid & 1) * 64;
  int n_wave = (wid >> 1) * 32;
  int kq = (tid & 7) * 4;          // B staging: k-block within wave
  int nq = ((tid >> 3) & 31) * 4;  // B staging: n-block

  v8f acc[4][2];
#pragma unroll
  for (int mt = 0; mt < 4; ++mt)
#pragma unroll
    for (int nt = 0; nt < 2; ++nt) acc[mt][nt] = {};

  // ---- prologue: fill buffer 0 ----
  if (tid < 32) {
    tdm_load_tile_bf16((unsigned)(size_t)&As[0][0], A + (size_t)m0 * K,
                       (unsigned)K, (unsigned)M, (unsigned long long)K,
                       32u, 128u);
  }
  gemm_stage_b(W, &Bs[0][0], 0, N, n0, kq, nq);
  if (tid < 32) __builtin_amdgcn_s_wait_tensorcnt(0);
  __syncthreads();

  int buf = 0;
  for (int kb = 0; kb < K; kb += 32) {
    int nxt = kb + 32;
    if (nxt < K) {
      // prefetch next tile into the other buffer (TDM overlaps compute)
      if (tid < 32) {
        tdm_load_tile_bf16((unsigned)(size_t)&As[buf ^ 1][0],
                           A + (size_t)m0 * K + nxt,
                           (unsigned)K, (unsigned)M, (unsigned long long)K,
                           32u, 128u);
      }
      gemm_stage_b(W, &Bs[buf ^ 1][0], nxt, N, n0, kq, nq);
      if (nxt + 32 < K)  // global_prefetch_b8 two tiles ahead
        __builtin_prefetch(&W[(size_t)(nxt + 32 + (tid & 7)) * N + n0 + nq], 0, 0);
    }
    // ---- compute on current buffer ----
    v16bf bfrag[2];
#pragma unroll
    for (int nt = 0; nt < 2; ++nt)
      bfrag[nt] = load_bT_frag(&Bs[buf][(size_t)(n_wave + nt * 16) * BS_LD], BS_LD);
#pragma unroll
    for (int mt = 0; mt < 4; ++mt) {
      v16bf afrag = load_a_frag(&As[buf][(size_t)(m_wave + mt * 16) * AS_LD], AS_LD);
#pragma unroll
      for (int nt = 0; nt < 2; ++nt)
        acc[mt][nt] = wmma_bf16(afrag, bfrag[nt], acc[mt][nt]);
    }
    if (nxt < K) {
      if (tid < 32) __builtin_amdgcn_s_wait_tensorcnt(0);
      __syncthreads();
      buf ^= 1;
    }
  }

  // ---- Epilogue ----
  int lane = tid & 31;
  int n = lane & 15, mb = (lane >> 4) << 3;
#pragma unroll
  for (int mt = 0; mt < 4; ++mt) {
#pragma unroll
    for (int nt = 0; nt < 2; ++nt) {
      int gcol = n0 + n_wave + nt * 16 + n;
      float bv = bias[gcol];
      int ms = m0 + m_wave + mt * 16;   // 16-row tiles never straddle a seq
      if (mode == 0) {
#pragma unroll
        for (int r = 0; r < 8; ++r)
          outF[(size_t)(ms + mb + r) * N + gcol] = acc[mt][nt][r] + bv;
      } else if (mode == 1) {
#pragma unroll
        for (int r = 0; r < 8; ++r) {
          float v = acc[mt][nt][r] + bv;
          v = 0.5f * v * (1.0f + erff(v * 0.70710678118f));
          outB[(size_t)(ms + mb + r) * N + gcol] = (bf16_t)v;
        }
      } else if (mode == 2) {  // [seq][h][t][d]
        int seq = ms / SLEN, tbase = ms - seq * SLEN + mb;
        int h = gcol >> 6, d = gcol & 63;
        bf16_t* dst = outB + (((size_t)seq * NH + h) * SLEN + tbase) * HD + d;
#pragma unroll
        for (int r = 0; r < 8; ++r) dst[(size_t)r * HD] = (bf16_t)(acc[mt][nt][r] + bv);
      } else {                  // mode 3: [seq][h][d][t], vector store along t
        int seq = ms / SLEN, tbase = ms - seq * SLEN + mb;
        int h = gcol >> 6, d = gcol & 63;
        union { uint4 u4; bf16_t hx[8]; } pk;
#pragma unroll
        for (int r = 0; r < 8; ++r) pk.hx[r] = (bf16_t)(acc[mt][nt][r] + bv);
        *(uint4*)(outB + (((size_t)seq * NH + h) * HD + d) * SLEN + tbase) = pk.u4;
      }
    }
  }
}

// ---------------------------------------------------------------------------
// Attention per (32-query chunk, head, seq).  Q,K packed [s][h][t][d];
// V packed transposed [s][h][d][t].  Scores+softmax in dynamic LDS.
// ---------------------------------------------------------------------------
static constexpr int ATT_QC  = 32;
static constexpr int ATT_SLD = 648;
static constexpr int ATT_SMEM =
    ATT_QC * ATT_SLD * 4 + ATT_QC * ATT_SLD * 2;

__global__ __launch_bounds__(256) void attn_kernel(
    const bf16_t* __restrict__ qh, const bf16_t* __restrict__ kh,
    const bf16_t* __restrict__ vt,
    const unsigned char* __restrict__ am1, const unsigned char* __restrict__ am2,
    bf16_t* __restrict__ ctxb) {
  extern __shared__ char smem[];
  float*  Sf = (float*)smem;                            // [32][648]
  bf16_t* Pb = (bf16_t*)(smem + ATT_QC * ATT_SLD * 4);  // [32][648]

  int qc = blockIdx.x, h = blockIdx.y, s = blockIdx.z;
  size_t hb = ((size_t)s * NH + h) * SLEN * HD;
  const bf16_t* Q  = qh + hb;
  const bf16_t* Kh = kh + hb;
  const bf16_t* Vt = vt + hb;  // [64][640]
  int t0 = qc * ATT_QC;

  int tid = threadIdx.x;
  int w = tid >> 5;
  int mrow = w & 1;
  int wcol = w >> 1;

  // ---- S = Q @ K^T ----
  v16bf aq0 = load_a_frag(Q + (size_t)(t0 + mrow * 16) * HD + 0,  HD);
  v16bf aq1 = load_a_frag(Q + (size_t)(t0 + mrow * 16) * HD + 32, HD);
#pragma unroll 1
  for (int j = 0; j < 10; ++j) {
    int ntile = wcol + 4 * j;
    v16bf b0 = load_bT_frag(Kh + (size_t)(ntile * 16) * HD + 0,  HD);
    v16bf b1 = load_bT_frag(Kh + (size_t)(ntile * 16) * HD + 32, HD);
    v8f acc = {};
    acc = wmma_bf16(aq0, b0, acc);
    acc = wmma_bf16(aq1, b1, acc);
    store_c_f32(&Sf[(mrow * 16) * ATT_SLD + ntile * 16], ATT_SLD, acc);
  }
  __syncthreads();

  // ---- masked softmax (8 lanes / row) ----
  int r = tid >> 3, sub = tid & 7;
  int qrow = t0 + r;
  const unsigned char* am =
      ((s & 1) ? am2 : am1) + ((size_t)(s >> 1) * SLEN + qrow) * SLEN;
  float mx = -3.0e38f;
  for (int c = sub; c < SLEN; c += 8) {
    float v = Sf[r * ATT_SLD + c] * 0.125f + (am[c] ? 0.0f : -10000.0f);
    Sf[r * ATT_SLD + c] = v;
    mx = fmaxf(mx, v);
  }
  mx = fmaxf(mx, __shfl_xor(mx, 1));
  mx = fmaxf(mx, __shfl_xor(mx, 2));
  mx = fmaxf(mx, __shfl_xor(mx, 4));
  float sum = 0.f;
  for (int c = sub; c < SLEN; c += 8) {
    float e = __expf(Sf[r * ATT_SLD + c] - mx);
    Sf[r * ATT_SLD + c] = e;
    sum += e;
  }
  sum += __shfl_xor(sum, 1);
  sum += __shfl_xor(sum, 2);
  sum += __shfl_xor(sum, 4);
  float inv = 1.0f / sum;
  for (int c = sub; c < SLEN; c += 8)
    Pb[r * ATT_SLD + c] = (bf16_t)(Sf[r * ATT_SLD + c] * inv);
  __syncthreads();

  // ---- ctx = P @ V  (V^T layout -> contiguous B fragments) ----
  v8f acc = {};
#pragma unroll 1
  for (int kk = 0; kk < 20; ++kk) {
    v16bf a = load_a_frag(&Pb[(mrow * 16) * ATT_SLD + kk * 32], ATT_SLD);
    v16bf b = load_bT_frag(Vt + (size_t)(wcol * 16) * SLEN + kk * 32, SLEN);
    acc = wmma_bf16(a, b, acc);
  }
  int lane = tid & 31;
  int n = lane & 15, mb = (lane >> 4) << 3;
  bf16_t* out = ctxb + (size_t)(s * SLEN + t0 + mrow * 16) * DM + h * HD + wcol * 16;
#pragma unroll
  for (int rr = 0; rr < 8; ++rr) out[(size_t)(mb + rr) * DM + n] = (bf16_t)acc[rr];
}

// ---------------------------------------------------------------------------
// x = LN(x + y); writes x_f32 and x_bf16.
// ---------------------------------------------------------------------------
__global__ __launch_bounds__(256) void resln_kernel(
    const float* __restrict__ x, const float* __restrict__ y,
    const float* __restrict__ g, const float* __restrict__ bta,
    float* __restrict__ xo, bf16_t* __restrict__ xbo) {
  __shared__ float red[256];
  __shared__ float mean_sh, rstd_sh;
  int row = blockIdx.x, tid = threadIdx.x;
  const float* xr = x + (size_t)row * DM;
  const float* yr = y + (size_t)row * DM;
  float e[3];
#pragma unroll
  for (int j = 0; j < 3; ++j) e[j] = xr[tid + j * 256] + yr[tid + j * 256];
  float sum = block_reduce_sum256(e[0] + e[1] + e[2], red);
  if (tid == 0) mean_sh = sum * (1.0f / (float)DM);
  __syncthreads();
  float m = mean_sh;
  float v0 = e[0] - m, v1 = e[1] - m, v2 = e[2] - m;
  float var = block_reduce_sum256(v0 * v0 + v1 * v1 + v2 * v2, red);
  if (tid == 0) rstd_sh = rsqrtf(var * (1.0f / (float)DM) + 1e-5f);
  __syncthreads();
  float rs = rstd_sh;
  float* xw = xo + (size_t)row * DM;
  bf16_t* xbr = xbo + (size_t)row * DM;
#pragma unroll
  for (int j = 0; j < 3; ++j) {
    int d = tid + j * 256;
    float val = (e[j] - m) * rs * g[d] + bta[d];
    xw[d] = val;
    xbr[d] = (bf16_t)val;
  }
}

// ---------------------------------------------------------------------------
// Head kernels
// ---------------------------------------------------------------------------
__global__ __launch_bounds__(256) void head_norm_kernel(
    const float* __restrict__ xf, float* __restrict__ on) {
  __shared__ float red[256];
  __shared__ float inv_sh;
  int s = blockIdx.x, tid = threadIdx.x;
  const float* xr = xf + (size_t)s * SLEN * DM;
  float e[3];
#pragma unroll
  for (int j = 0; j < 3; ++j) e[j] = xr[tid + j * 256];
  float ss = block_reduce_sum256(e[0]*e[0] + e[1]*e[1] + e[2]*e[2], red);
  if (tid == 0) inv_sh = 1.0f / fmaxf(sqrtf(ss), 1e-12f);
  __syncthreads();
  float inv = inv_sh;
#pragma unroll
  for (int j = 0; j < 3; ++j) on[(size_t)s * DM + tid + j * 256] = e[j] * inv;
}

__global__ __launch_bounds__(256) void head_tanh_kernel(
    const float* __restrict__ on, const float* __restrict__ Wd,
    const float* __restrict__ bd, float* __restrict__ tvec) {
  int i = blockIdx.x, tid = threadIdx.x;
  const float* c = on + (size_t)(2 * i) * DM;
#pragma unroll 1
  for (int j = 0; j < 3; ++j) {
    int col = tid + j * 256;
    float acc = bd[col];
    for (int k = 0; k < DM; ++k) acc += c[k] * Wd[(size_t)k * DM + col];
    tvec[(size_t)i * DM + col] = tanhf(acc);
  }
}

__global__ __launch_bounds__(32) void head_final_kernel(
    const float* __restrict__ tvec, const float* __restrict__ Wcls,
    const float* __restrict__ bcls, const int* __restrict__ labels,
    float* __restrict__ out) {
  __shared__ float loss_sh[8];
  __shared__ float prob_sh[8];
  int tid = threadIdx.x;
  if (tid < 8) {
    const float* tv = tvec + (size_t)tid * DM;
    float l0 = bcls[0], l1 = bcls[1];
    for (int k = 0; k < DM; ++k) {
      l0 += tv[k] * Wcls[k * 2 + 0];
      l1 += tv[k] * Wcls[k * 2 + 1];
    }
    float mx = fmaxf(l0, l1);
    float lse = mx + logf(__expf(l0 - mx) + __expf(l1 - mx));
    int lab = labels[tid];
    float lp = (lab == 0 ? l0 : l1) - lse;
    loss_sh[tid] = -lp;
    prob_sh[tid] = (l0 > l1) ? 0.0f : 1.0f;
  }
  __syncthreads();
  if (tid == 0) {
    float acc = 0.f;
    for (int i = 0; i < 8; ++i) acc += loss_sh[i];
    out[0] = (acc / 8.0f) * 0.9f;  // * (1 - LAMBD)
    for (int i = 0; i < 8; ++i) out[1 + i] = prob_sh[i];
  }
}

// ---------------------------------------------------------------------------
// Host-side orchestration
// ---------------------------------------------------------------------------
static inline char* bump(char*& p, size_t bytes) {
  char* r = p;
  p += (bytes + 255) & ~(size_t)255;
  return r;
}

extern "C" void kernel_launch(void* const* d_in, const int* in_sizes, int n_in,
                              void* d_out, int out_size, void* d_ws, size_t ws_size,
                              hipStream_t stream) {
  (void)in_sizes; (void)n_in; (void)out_size; (void)ws_size;
  const int*  ids1   = (const int*)d_in[1];
  const int*  ids2   = (const int*)d_in[2];
  const int*  pos1   = (const int*)d_in[3];
  const int*  pos2   = (const int*)d_in[4];
  const unsigned char* am1 = (const unsigned char*)d_in[5];
  const unsigned char* am2 = (const unsigned char*)d_in[6];
  const int*  labels = (const int*)d_in[9];
  const float* word_emb = (const float*)d_in[10];
  const float* pos_emb  = (const float*)d_in[11];
  const float* type_emb = (const float*)d_in[12];
  const float* emb_g = (const float*)d_in[13];
  const float* emb_b = (const float*)d_in[14];
  const float* Wq = (const float*)d_in[15]; const float* bq = (const float*)d_in[16];
  const float* Wk = (const float*)d_in[17]; const float* bk = (const float*)d_in[18];
  const float* Wv = (const float*)d_in[19]; const float* bv = (const float*)d_in[20];
  const float* Wo = (const float*)d_in[21]; const float* bo = (const float*)d_in[22];
  const float* ln1g = (const float*)d_in[23]; const float* ln1b = (const float*)d_in[24];
  const float* W1 = (const float*)d_in[25]; const float* b1 = (const float*)d_in[26];
  const float* W2 = (const float*)d_in[27]; const float* b2 = (const float*)d_in[28];
  const float* ln2g = (const float*)d_in[29]; const float* ln2b = (const float*)d_in[30];
  const float* Wd   = (const float*)d_in[31]; const float* bd = (const float*)d_in[32];
  const float* Wcls = (const float*)d_in[33]; const float* bcls = (const float*)d_in[34];
  float* out = (float*)d_out;

  const size_t nDD = (size_t)4 * DM * DM;
  const size_t nDF = (size_t)4 * DM * DF;

  char* p = (char*)d_ws;
  bf16_t* wqb = (bf16_t*)bump(p, nDD * 2);
  bf16_t* wkb = (bf16_t*)bump(p, nDD * 2);
  bf16_t* wvb = (bf16_t*)bump(p, nDD * 2);
  bf16_t* wob = (bf16_t*)bump(p, nDD * 2);
  bf16_t* w1b = (bf16_t*)bump(p, nDF * 2);
  bf16_t* w2b = (bf16_t*)bump(p, nDF * 2);
  float*  xf  = (float*) bump(p, (size_t)MROWS * DM * 4);
  bf16_t* xb  = (bf16_t*)bump(p, (size_t)MROWS * DM * 2);
  float*  tmp = (float*) bump(p, (size_t)MROWS * DM * 4);
  bf16_t* qhb = (bf16_t*)bump(p, (size_t)MROWS * DM * 2);
  bf16_t* khb = (bf16_t*)bump(p, (size_t)MROWS * DM * 2);
  bf16_t* vtb = (bf16_t*)bump(p, (size_t)MROWS * DM * 2);
  bf16_t* ctx = (bf16_t*)bump(p, (size_t)MROWS * DM * 2);
  bf16_t* hmb = (bf16_t*)bump(p, (size_t)MROWS * DF * 2);
  float*  onb = (float*) bump(p, (size_t)SEQS * DM * 4);
  float*  tvb = (float*) bump(p, (size_t)8 * DM * 4);

  auto cvt = [&](const float* s, bf16_t* d, size_t n) {
    convert_bf16_kernel<<<(unsigned)((n + 255) / 256), 256, 0, stream>>>(s, d, (long long)n);
  };
  cvt(Wq, wqb, nDD); cvt(Wk, wkb, nDD); cvt(Wv, wvb, nDD); cvt(Wo, wob, nDD);
  cvt(W1, w1b, nDF); cvt(W2, w2b, nDF);

  embed_kernel<<<MROWS, 256, 0, stream>>>(ids1, ids2, pos1, pos2, am1, am2,
      word_emb, pos_emb, type_emb, emb_g, emb_b, xf, xb);

  const dim3 gProj(DM / 128, MROWS / 128);   // (6, 80)
  const dim3 gFF1(DF / 128, MROWS / 128);    // (24, 80)

  for (int i = 0; i < 4; ++i) {
    const size_t oDD = (size_t)i * DM * DM;
    const size_t oDF = (size_t)i * DM * DF;
    gemm_bf16_kernel<<<gProj, 256, 0, stream>>>(xb, wqb + oDD, bq + i * DM,
        nullptr, qhb, MROWS, DM, DM, 2);
    gemm_bf16_kernel<<<gProj, 256, 0, stream>>>(xb, wkb + oDD, bk + i * DM,
        nullptr, khb, MROWS, DM, DM, 2);
    gemm_bf16_kernel<<<gProj, 256, 0, stream>>>(xb, wvb + oDD, bv + i * DM,
        nullptr, vtb, MROWS, DM, DM, 3);
    attn_kernel<<<dim3(SLEN / ATT_QC, NH, SEQS), 256, ATT_SMEM, stream>>>(
        qhb, khb, vtb, am1, am2, ctx);
    gemm_bf16_kernel<<<gProj, 256, 0, stream>>>(ctx, wob + oDD, bo + i * DM,
        tmp, nullptr, MROWS, DM, DM, 0);
    resln_kernel<<<MROWS, 256, 0, stream>>>(xf, tmp, ln1g + i * DM, ln1b + i * DM, xf, xb);
    gemm_bf16_kernel<<<gFF1, 256, 0, stream>>>(xb, w1b + oDF, b1 + i * DF,
        nullptr, hmb, MROWS, DF, DM, 1);
    gemm_bf16_kernel<<<gProj, 256, 0, stream>>>(hmb, w2b + oDF, b2 + i * DM,
        tmp, nullptr, MROWS, DM, DF, 0);
    resln_kernel<<<MROWS, 256, 0, stream>>>(xf, tmp, ln2g + i * DM, ln2b + i * DM, xf, xb);
  }

  head_norm_kernel<<<SEQS, 256, 0, stream>>>(xf, onb);
  head_tanh_kernel<<<8, 256, 0, stream>>>(onb, Wd, bd, tvb);
  head_final_kernel<<<1, 32, 0, stream>>>(tvb, Wcls, bcls, labels, out);
}
